// Block_19713899888821
// MI455X (gfx1250) — compile-verified, matrix operand order checked by reading, not run on hardware
//
#include <hip/hip_runtime.h>
#include <math.h>

// ---------------------------------------------------------------------------
// Types for CDNA5 WMMA (wave32): 16x16x32 bf16 -> f32
// ---------------------------------------------------------------------------
typedef __bf16 bf16;
typedef __bf16 bf16x4 __attribute__((ext_vector_type(4)));
typedef __bf16 bf16x8 __attribute__((ext_vector_type(8)));
typedef __bf16 v16bf  __attribute__((ext_vector_type(16)));
typedef float  v8f    __attribute__((ext_vector_type(8)));

__device__ __forceinline__ bf16 f2bf(float f) {
  union { float f; unsigned u; } a; a.f = f;
  unsigned r = a.u + 0x7FFFu + ((a.u >> 16) & 1u);   // round-to-nearest-even
  union { unsigned short s; bf16 b; } o;
  o.s = (unsigned short)(r >> 16);
  return o.b;
}

// Fragment load per ISA 16-bit A/B layout: lane holds 8 contiguous K at +0 and
// 8 contiguous K at +16 elements (two 16-byte ds_load_b128).
__device__ __forceinline__ v16bf load_frag(const bf16* p) {
  bf16x8 lo = *(const bf16x8*)(p);
  bf16x8 hi = *(const bf16x8*)(p + 16);
  return __builtin_shufflevector(lo, hi, 0,1,2,3,4,5,6,7,8,9,10,11,12,13,14,15);
}

__device__ __forceinline__ v8f wmma_bf16(v16bf a, v16bf b, v8f c) {
  return __builtin_amdgcn_wmma_f32_16x16x32_bf16(false, a, false, b, (short)0, c,
                                                 false, false);
}

// ---------------------------------------------------------------------------
// 16-byte global->LDS copy. Prefer the CDNA5 async-to-LDS path (ASYNCcnt)
// when the toolchain declares the builtin; otherwise plain b128 load+store.
// Builtin signature (per hipcc diagnostic): param0 = i32x4 addrspace(1)*,
// param1 = i32x4 addrspace(3)*, then two imm ints (offset, cpol).
// ---------------------------------------------------------------------------
#if __has_builtin(__builtin_amdgcn_global_load_async_to_lds_b128)
#define ASYNC_LDS 1
#endif

typedef int i32x4 __attribute__((ext_vector_type(4)));
typedef __attribute__((address_space(1))) i32x4 gl_i32x4;
typedef __attribute__((address_space(3))) i32x4 lds_i32x4;

__device__ __forceinline__ void cp16(void* lds, const void* g) {
#ifdef ASYNC_LDS
  __builtin_amdgcn_global_load_async_to_lds_b128(
      (gl_i32x4*)g, (lds_i32x4*)lds, 0, 0);
#else
  *(uint4*)lds = *(const uint4*)g;
#endif
}

__device__ __forceinline__ void async_wait() {
#ifdef ASYNC_LDS
#if __has_builtin(__builtin_amdgcn_s_wait_asynccnt)
  __builtin_amdgcn_s_wait_asynccnt(0);
#else
  asm volatile("s_wait_asynccnt 0x0" ::: "memory");
#endif
#endif
}

__device__ __forceinline__ void store_out(float* p, float v) { *p = v; }
__device__ __forceinline__ void store_out(bf16*  p, float v) { *p = f2bf(v); }

// ---------------------------------------------------------------------------
// Weight transpose + convert: W[K][N] f32 -> Wt[N][K] bf16
// ---------------------------------------------------------------------------
__global__ __launch_bounds__(256)
void wtr_kernel(const float* __restrict__ W, bf16* __restrict__ Wt,
                int K, int N) {
  size_t idx = (size_t)blockIdx.x * 256 + threadIdx.x;   // over N*K
  int k = (int)(idx % K);
  int n = (int)(idx / K);
  Wt[idx] = f2bf(W[(size_t)k * N + n]);
}

// ---------------------------------------------------------------------------
// V transpose: qkv bf16 [B*T][3C] -> vT[b][h][d][t] bf16
// ---------------------------------------------------------------------------
__global__ __launch_bounds__(256)
void vtrans_kernel(const bf16* __restrict__ qkv, bf16* __restrict__ vT) {
  size_t idx = (size_t)blockIdx.x * 256 + threadIdx.x;   // 32*64*2048 elements
  int t  = (int)(idx & 2047);
  int d  = (int)((idx >> 11) & 63);
  int bh = (int)(idx >> 17);                 // 0..31
  int bz = bh >> 4, h = bh & 15;
  vT[idx] = qkv[((size_t)bz * 2048 + t) * 3072 + 2048 + h * 64 + d];
}

// ---------------------------------------------------------------------------
// LayerNorm: one block (256 threads) per row of 1024; bf16 output
// ---------------------------------------------------------------------------
__global__ __launch_bounds__(256)
void ln_kernel(const float* __restrict__ x, const float* __restrict__ g,
               const float* __restrict__ b, bf16* __restrict__ out) {
  const int row = blockIdx.x;
  const float* xr = x + (size_t)row * 1024;
  const int c = threadIdx.x * 4;
  float4 v = *(const float4*)(xr + c);
  float s  = v.x + v.y + v.z + v.w;
  float s2 = v.x*v.x + v.y*v.y + v.z*v.z + v.w*v.w;
  #pragma unroll
  for (int off = 16; off > 0; off >>= 1) {
    s  += __shfl_xor(s, off);
    s2 += __shfl_xor(s2, off);
  }
  __shared__ float ws[8], ws2[8];
  const int wv = threadIdx.x >> 5, ln = threadIdx.x & 31;
  if (ln == 0) { ws[wv] = s; ws2[wv] = s2; }
  __syncthreads();
  if (wv == 0) {
    s  = (ln < 8) ? ws[ln]  : 0.f;
    s2 = (ln < 8) ? ws2[ln] : 0.f;
    #pragma unroll
    for (int off = 4; off > 0; off >>= 1) {
      s  += __shfl_xor(s, off);
      s2 += __shfl_xor(s2, off);
    }
    if (ln == 0) { ws[0] = s; ws2[0] = s2; }
  }
  __syncthreads();
  const float mu  = ws[0] * (1.f / 1024.f);
  const float var = ws2[0] * (1.f / 1024.f) - mu * mu;
  const float rstd = rsqrtf(var + 1e-5f);
  float4 gg = *(const float4*)(g + c);
  float4 bb = *(const float4*)(b + c);
  bf16x4 o;
  o[0] = f2bf((v.x - mu) * rstd * gg.x + bb.x);
  o[1] = f2bf((v.y - mu) * rstd * gg.y + bb.y);
  o[2] = f2bf((v.z - mu) * rstd * gg.z + bb.z);
  o[3] = f2bf((v.w - mu) * rstd * gg.w + bb.w);
  *(bf16x4*)(out + (size_t)row * 1024 + c) = o;
}

// ---------------------------------------------------------------------------
// Tiled GEMM: out[M,N] = A[M,K](bf16) * Wt[N,K](bf16)^T  (+bias)(+gelu)(+res)
// 128x128 tile per block, 256 threads = 8 waves (4 M-slabs x 2 N-slabs);
// each wave: 32 rows x 64 cols = 2x4 WMMA accumulators. Double-buffered LDS,
// async global->LDS staging. EPI bit0=bias, bit1=gelu(tanh), bit2=residual.
// ---------------------------------------------------------------------------
template<int EPI, typename OutT>
__global__ __launch_bounds__(256)
void gemm_kernel(const bf16* __restrict__ A, const bf16* __restrict__ Wt,
                 const float* __restrict__ bias, const float* __restrict__ res,
                 OutT* __restrict__ out, int N, int K) {
  __shared__ bf16 Ab[2][128 * 32];   // [m][k]
  __shared__ bf16 Bb[2][128 * 32];   // [n][k]
  const int tid  = threadIdx.x;
  const int wv   = tid >> 5;
  const int ln   = tid & 31;
  const int hl   = ln >> 4;           // half-wave: K-offset +0 / +8
  const int l16  = ln & 15;
  const int m_base = blockIdx.y * 128;
  const int n_base = blockIdx.x * 128;
  const int wm = (wv >> 1) * 32;      // wave M-slab within block
  const int wn = (wv & 1) * 64;       // wave N-slab within block

  v8f acc[2][4];
  #pragma unroll
  for (int mi = 0; mi < 2; ++mi)
    #pragma unroll
    for (int t = 0; t < 4; ++t)
      #pragma unroll
      for (int i = 0; i < 8; ++i) acc[mi][t][i] = 0.f;

  const int nk = K >> 5;

  // copy one 128x32 bf16 tile pair (512 x 16B chunks each, 2 per thread)
  auto copy_tiles = [&](int buf, int k0) {
    #pragma unroll
    for (int i = 0; i < 2; ++i) {
      const int c = tid * 2 + i;
      const int r = c >> 2;
      const int off = (c & 3) * 8;
      cp16(&Ab[buf][r * 32 + off], A  + (size_t)(m_base + r) * K + k0 + off);
      cp16(&Bb[buf][r * 32 + off], Wt + (size_t)(n_base + r) * K + k0 + off);
    }
  };

  copy_tiles(0, 0);
  for (int kt = 0; kt < nk; ++kt) {
    async_wait();
    __syncthreads();
    if (kt + 1 < nk) copy_tiles((kt + 1) & 1, (kt + 1) * 32);
    const bf16* As = Ab[kt & 1];
    const bf16* Bs = Bb[kt & 1];
    v16bf a0 = load_frag(&As[(wm + l16) * 32 + hl * 8]);
    v16bf a1 = load_frag(&As[(wm + 16 + l16) * 32 + hl * 8]);
    #pragma unroll
    for (int t = 0; t < 4; ++t) {
      v16bf b = load_frag(&Bs[(wn + t * 16 + l16) * 32 + hl * 8]);
      acc[0][t] = wmma_bf16(a0, b, acc[0][t]);
      acc[1][t] = wmma_bf16(a1, b, acc[1][t]);
    }
  }

  // Epilogue. C/D layout: lane -> col n = t*16+(l&15); elem v -> row (l>=16?8:0)+v.
  #pragma unroll
  for (int t = 0; t < 4; ++t) {
    const int gn = n_base + wn + t * 16 + l16;
    float bv = 0.f;
    if constexpr (EPI & 1) bv = bias[gn];
    #pragma unroll
    for (int mi = 0; mi < 2; ++mi) {
      #pragma unroll
      for (int v = 0; v < 8; ++v) {
        const int gm = m_base + wm + mi * 16 + hl * 8 + v;
        float xv = acc[mi][t][v] + bv;
        if constexpr (EPI & 2) {
          const float u = 0.7978845608028654f * (xv + 0.044715f * xv * xv * xv);
          xv = 0.5f * xv * (1.f + tanhf(u));
        }
        if constexpr (EPI & 4) xv += res[(size_t)gm * N + gn];
        store_out(&out[(size_t)gm * N + gn], xv);
      }
    }
  }
}

// ---------------------------------------------------------------------------
// Flash attention, causal. One block = (batch, head, 64-query tile).
// 128 threads = 4 waves; wave owns 16 query rows. All staging is contiguous
// bf16 copies (V comes pre-transposed from vT).
// ---------------------------------------------------------------------------
__global__ __launch_bounds__(128)
void attn_kernel(const bf16* __restrict__ qkv, const bf16* __restrict__ vT,
                 bf16* __restrict__ y) {
  const int T = 2048, C3 = 3072;
  const int qt = blockIdx.x;
  const int h  = blockIdx.y;
  const int bz = blockIdx.z;
  const int tid = threadIdx.x, wv = tid >> 5, ln = tid & 31;
  const int hl = ln >> 4, l16 = ln & 15;
  const int q0 = qt * 64;

  const bf16* qbase = qkv + (size_t)bz * T * C3 + h * 64;
  const bf16* kbase = qbase + 1024;
  const bf16* vbase = vT + ((size_t)bz * 16 + h) * 64 * 2048;   // [d][t]

  __shared__ bf16 Qb[64 * 64];   // [qrow][d]
  __shared__ bf16 Kb[64 * 64];   // [key][d]
  __shared__ bf16 Vt[64 * 64];   // [d][key]
  __shared__ bf16 Pb[64 * 64];   // [qrow][key] (wave-private 16-row slabs)

  // Q tile: 512 x 16B chunks, 4 per thread
  #pragma unroll
  for (int i = 0; i < 4; ++i) {
    const int c = tid * 4 + i;
    const int r = c >> 3;
    const int off = (c & 7) * 8;
    cp16(&Qb[r * 64 + off], qbase + (size_t)(q0 + r) * C3 + off);
  }

  v8f o[4];
  float row_m[8], row_l[8];
  #pragma unroll
  for (int t = 0; t < 4; ++t)
    #pragma unroll
    for (int v = 0; v < 8; ++v) o[t][v] = 0.f;
  #pragma unroll
  for (int v = 0; v < 8; ++v) { row_m[v] = -INFINITY; row_l[v] = 0.f; }

  for (int j = 0; j <= qt; ++j) {
    __syncthreads();   // previous iteration's compute done before overwrite
    #pragma unroll
    for (int i = 0; i < 4; ++i) {
      const int c = tid * 4 + i;
      const int r = c >> 3;            // key row / d row
      const int off = (c & 7) * 8;
      cp16(&Kb[r * 64 + off], kbase + (size_t)(j * 64 + r) * C3 + off);
      cp16(&Vt[r * 64 + off], vbase + (size_t)r * 2048 + j * 64 + off);
    }
    async_wait();
    __syncthreads();

    // S = Q K^T for this wave's 16 rows (d=64 -> 2 k-steps)
    v8f s[4];
    #pragma unroll
    for (int t = 0; t < 4; ++t)
      #pragma unroll
      for (int v = 0; v < 8; ++v) s[t][v] = 0.f;
    #pragma unroll
    for (int ks = 0; ks < 64; ks += 32) {
      v16bf a = load_frag(&Qb[(wv * 16 + l16) * 64 + ks + hl * 8]);
      #pragma unroll
      for (int t = 0; t < 4; ++t) {
        v16bf b = load_frag(&Kb[(t * 16 + l16) * 64 + ks + hl * 8]);
        s[t] = wmma_bf16(a, b, s[t]);
      }
    }

    // scale + causal mask + online softmax update
    const int keyb = j * 64;
    #pragma unroll
    for (int v = 0; v < 8; ++v) {
      const int qg = q0 + wv * 16 + hl * 8 + v;
      float mx = -INFINITY;
      #pragma unroll
      for (int t = 0; t < 4; ++t) {
        const int kg = keyb + t * 16 + l16;
        float val = s[t][v] * 0.125f;             // 1/sqrt(64)
        if (kg > qg) val = -INFINITY;
        s[t][v] = val;
        mx = fmaxf(mx, val);
      }
      #pragma unroll
      for (int off = 1; off < 16; off <<= 1) mx = fmaxf(mx, __shfl_xor(mx, off));
      const float nm = fmaxf(row_m[v], mx);
      const float sc = __expf(row_m[v] - nm);
      float ts = 0.f;
      #pragma unroll
      for (int t = 0; t < 4; ++t) {
        const float p = __expf(s[t][v] - nm);
        s[t][v] = p;
        ts += p;
      }
      #pragma unroll
      for (int off = 1; off < 16; off <<= 1) ts += __shfl_xor(ts, off);
      row_l[v] = row_l[v] * sc + ts;
      row_m[v] = nm;
      #pragma unroll
      for (int t = 0; t < 4; ++t) o[t][v] *= sc;
    }

    // Spill P (C/D layout) -> Pb (A-layout source), wave-private rows
    #pragma unroll
    for (int t = 0; t < 4; ++t)
      #pragma unroll
      for (int v = 0; v < 8; ++v)
        Pb[(wv * 16 + hl * 8 + v) * 64 + t * 16 + l16] = f2bf(s[t][v]);

    // O += P * V (64 keys -> 2 k-steps)
    #pragma unroll
    for (int ks = 0; ks < 64; ks += 32) {
      v16bf a = load_frag(&Pb[(wv * 16 + l16) * 64 + ks + hl * 8]);
      #pragma unroll
      for (int t = 0; t < 4; ++t) {
        v16bf b = load_frag(&Vt[(t * 16 + l16) * 64 + ks + hl * 8]);
        o[t] = wmma_bf16(a, b, o[t]);
      }
    }
  }

  // Finalize: O / l, write y[b, q, h*64 + d] (bf16)
  #pragma unroll
  for (int t = 0; t < 4; ++t) {
    const int d = t * 16 + l16;
    #pragma unroll
    for (int v = 0; v < 8; ++v) {
      const int qg = q0 + wv * 16 + hl * 8 + v;
      y[((size_t)bz * T + qg) * 1024 + h * 64 + d] = f2bf(o[t][v] / row_l[v]);
    }
  }
}

// ---------------------------------------------------------------------------
// Host launcher
// ---------------------------------------------------------------------------
extern "C" void kernel_launch(void* const* d_in, const int* in_sizes, int n_in,
                              void* d_out, int out_size, void* d_ws, size_t ws_size,
                              hipStream_t stream) {
  (void)in_sizes; (void)n_in; (void)out_size;
  const float* x      = (const float*)d_in[0];
  const float* ln1_g  = (const float*)d_in[1];
  const float* ln1_b  = (const float*)d_in[2];
  const float* W_attn = (const float*)d_in[3];
  const float* b_attn = (const float*)d_in[4];
  const float* W_o    = (const float*)d_in[5];
  const float* b_o    = (const float*)d_in[6];
  const float* ln2_g  = (const float*)d_in[7];
  const float* ln2_b  = (const float*)d_in[8];
  const float* W_fc   = (const float*)d_in[9];
  const float* b_fc   = (const float*)d_in[10];
  const float* W_fc2  = (const float*)d_in[11];
  const float* b_fc2  = (const float*)d_in[12];
  float* out = (float*)d_out;

  const size_t M = 4096;   // B*T
  char* w = (char*)d_ws;
  bf16*  ln_out = (bf16*)w;  w += M * 1024 * 2;
  bf16*  qkv    = (bf16*)w;  w += M * 3072 * 2;
  bf16*  vT     = (bf16*)w;  w += (size_t)32 * 64 * 2048 * 2;
  bf16*  yb     = (bf16*)w;  w += M * 1024 * 2;
  float* x1     = (float*)w; w += M * 1024 * 4;
  bf16*  a1     = (bf16*)w;  w += M * 4096 * 2;
  bf16*  WtA    = (bf16*)w;  w += (size_t)1024 * 3072 * 2;
  bf16*  WtO    = (bf16*)w;  w += (size_t)1024 * 1024 * 2;
  bf16*  WtF    = (bf16*)w;  w += (size_t)1024 * 4096 * 2;
  bf16*  WtF2   = (bf16*)w;  w += (size_t)4096 * 1024 * 2;
  if (ws_size < (size_t)(w - (char*)d_ws)) return;

  // Weight convert+transpose (bf16 [N][K])
  wtr_kernel<<<dim3(1024 * 3072 / 256), dim3(256), 0, stream>>>(W_attn, WtA, 1024, 3072);
  wtr_kernel<<<dim3(1024 * 1024 / 256), dim3(256), 0, stream>>>(W_o,    WtO, 1024, 1024);
  wtr_kernel<<<dim3(1024 * 4096 / 256), dim3(256), 0, stream>>>(W_fc,   WtF, 1024, 4096);
  wtr_kernel<<<dim3(4096 * 1024 / 256), dim3(256), 0, stream>>>(W_fc2,  WtF2, 4096, 1024);

  // 1. LN1 -> bf16
  ln_kernel<<<dim3(4096), dim3(256), 0, stream>>>(x, ln1_g, ln1_b, ln_out);
  // 2. QKV = LN1 @ W_attn + b_attn  (bf16 out)
  gemm_kernel<1, bf16><<<dim3(3072 / 128, 4096 / 128), dim3(256), 0, stream>>>(
      ln_out, WtA, b_attn, nullptr, qkv, 3072, 1024);
  // 3. V transpose per head
  vtrans_kernel<<<dim3(32 * 64 * 2048 / 256), dim3(256), 0, stream>>>(qkv, vT);
  // 4. Causal flash attention (bf16 out)
  attn_kernel<<<dim3(32, 16, 2), dim3(128), 0, stream>>>(qkv, vT, yb);
  // 5. x1 = x + y @ W_o + b_o  (f32 out)
  gemm_kernel<5, float><<<dim3(1024 / 128, 4096 / 128), dim3(256), 0, stream>>>(
      yb, WtO, b_o, x, x1, 1024, 1024);
  // 6. LN2 -> bf16
  ln_kernel<<<dim3(4096), dim3(256), 0, stream>>>(x1, ln2_g, ln2_b, ln_out);
  // 7. a1 = gelu(LN2 @ W_fc + b_fc)  (bf16 out)
  gemm_kernel<3, bf16><<<dim3(4096 / 128, 4096 / 128), dim3(256), 0, stream>>>(
      ln_out, WtF, b_fc, nullptr, a1, 4096, 1024);
  // 8. out = x1 + a1 @ W_fc2 + b_fc2  (f32 out)
  gemm_kernel<5, float><<<dim3(1024 / 128, 4096 / 128), dim3(256), 0, stream>>>(
      a1, WtF2, b_fc2, x1, out, 1024, 4096);
}
